// BiMamba3Block_79370995630842
// MI455X (gfx1250) — matrix operand order repaired and checked by reading, not compiled, not run on hardware
//
#include <hip/hip_runtime.h>
#include <math.h>

// ---------------------------------------------------------------- constants
#define B_     2
#define L_     1024
#define DM     1024      // d_model
#define DI     2048      // d_inner
#define NH     32        // heads
#define HD     64        // head dim
#define DS     64        // d_state
#define RK     4         // rank
#define DPROJ  4640      // 2*DI + 2*RK*DS + NH
#define DPROJP 4672      // DPROJ padded to multiple of 64 (GEMM tile width)
#define DFF    4096
#define KC     4
#define MROWS  (B_*L_)   // 2048

typedef __attribute__((ext_vector_type(16))) __bf16 v16bf;
typedef __attribute__((ext_vector_type(8)))  __bf16 v8bf;
typedef __attribute__((ext_vector_type(8)))  float  v8f;

union ABf { v16bf v; v8bf h[2]; };

__device__ __forceinline__ unsigned short f2bf(float f) {
  unsigned int u = __float_as_uint(f);
  unsigned int lsb = (u >> 16) & 1u;
  u += 0x7fffu + lsb;               // round-to-nearest-even
  return (unsigned short)(u >> 16);
}

__device__ __forceinline__ float siluf(float x) { return x / (1.f + expf(-x)); }
__device__ __forceinline__ float sigf (float x) { return 1.f / (1.f + expf(-x)); }

// CDNA5 async global->LDS load (ASYNCcnt-tracked); vdst = per-lane LDS byte
// address, vaddr = 64-bit global address (GV mode, SADDR=off).
__device__ __forceinline__ void async_ld_b32(unsigned lds_off,
                                             const float* __restrict__ g) {
  asm volatile("global_load_async_to_lds_b32 %0, %1, off"
               :: "v"(lds_off), "v"((unsigned long long)(size_t)g)
               : "memory");
}

// ---------------------------------------------------------------- WMMA GEMM
// C[M,N] (f32, +optional resid) = A[M,K](bf16 row-major) * W[N,K](bf16 row-major)^T
// Block: 128 threads = 4 waves stacked in M; each wave owns a 64x64 output tile
// (4x4 WMMA fragments). Branchless: requires M%256==0, N%64==0, K%64==0.
// K-loop is software-pipelined (register ping-pong) + global_prefetch ahead.

__device__ __forceinline__ void load_frags(const unsigned short* const* Arow,
                                           const unsigned short* const* Wrow,
                                           int k0, ABf* af, v16bf* bf) {
  #pragma unroll
  for (int mi = 0; mi < 4; mi++) {
    af[mi].h[0] = *(const v8bf*)(Arow[mi] + k0);
    af[mi].h[1] = *(const v8bf*)(Arow[mi] + k0 + 8);
  }
  #pragma unroll
  for (int ni = 0; ni < 4; ni++)
    bf[ni] = *(const v16bf*)(Wrow[ni] + k0);
}

__device__ __forceinline__ void do_wmma(const ABf* af, const v16bf* bf,
                                        v8f acc[4][4]) {
  #pragma unroll
  for (int mi = 0; mi < 4; mi++)
    #pragma unroll
    for (int ni = 0; ni < 4; ni++)
      acc[mi][ni] = __builtin_amdgcn_wmma_f32_16x16x32_bf16(
          false, af[mi].v, false, bf[ni], (short)0, acc[mi][ni], false, false);
}

template <bool HASRES>
__global__ __launch_bounds__(128)
void gemm_bf16_wmma(const unsigned short* __restrict__ A,
                    const unsigned short* __restrict__ W,
                    float* __restrict__ C,
                    const float* __restrict__ resid,
                    int M, int N, int K)
{
  const int lane  = threadIdx.x & 31;
  const int wave  = threadIdx.x >> 5;
  const int mBase = blockIdx.y * 256 + wave * 64;
  const int nBase = blockIdx.x * 64;
  const int lr = lane & 15;   // A row-in-tile / B,C column-in-tile
  const int lh = lane >> 4;   // half-K select

  v8f acc[4][4];
  #pragma unroll
  for (int mi = 0; mi < 4; mi++)
    #pragma unroll
    for (int ni = 0; ni < 4; ni++)
      #pragma unroll
      for (int e = 0; e < 8; e++) acc[mi][ni][e] = 0.f;

  // A frags (this code's K order, same as B): lanes 0-15 hold K=0..15,
  // lanes 16-31 hold K=16..31 of row M=lane&15.
  const unsigned short* Arow[4];
  #pragma unroll
  for (int mi = 0; mi < 4; mi++)
    Arow[mi] = A + (size_t)(mBase + mi * 16 + lr) * K + lh * 16;

  // B frags (same K order as A): lanes 0-15 hold K=0..15 of column N=lane&15,
  // lanes 16-31 hold K=16..31.
  const unsigned short* Wrow[4];
  #pragma unroll
  for (int ni = 0; ni < 4; ni++)
    Wrow[ni] = W + (size_t)(nBase + ni * 16 + lr) * K + lh * 16;

  ABf  a0[4], a1[4];
  v16bf b0[4], b1[4];

  load_frags(Arow, Wrow, 0, a0, b0);
  for (int k0 = 0; k0 < K - 64; k0 += 64) {
    // prefetch one cacheline-stream step ahead of the pipeline
    #pragma unroll
    for (int mi = 0; mi < 4; mi++) __builtin_prefetch(Arow[mi] + k0 + 256, 0, 3);
    #pragma unroll
    for (int ni = 0; ni < 4; ni++) __builtin_prefetch(Wrow[ni] + k0 + 256, 0, 3);

    load_frags(Arow, Wrow, k0 + 32, a1, b1);
    do_wmma(a0, b0, acc);
    load_frags(Arow, Wrow, k0 + 64, a0, b0);
    do_wmma(a1, b1, acc);
  }
  load_frags(Arow, Wrow, K - 32, a1, b1);
  do_wmma(a0, b0, acc);
  do_wmma(a1, b1, acc);

  // C/D layout: vgpr r -> M = r + 8*(lane>>4); N = lane&15.
  #pragma unroll
  for (int mi = 0; mi < 4; mi++)
    #pragma unroll
    for (int ni = 0; ni < 4; ni++) {
      int col = nBase + ni * 16 + lr;
      #pragma unroll
      for (int r = 0; r < 8; r++) {
        int row = mBase + mi * 16 + lh * 8 + r;
        size_t idx = (size_t)row * N + col;
        float v = acc[mi][ni][r];
        if (HASRES) v += resid[idx];
        C[idx] = v;
      }
    }
}

// ---------------------------------------------------------------- elementwise
__global__ void cvt_bf16_kernel(const float* __restrict__ src,
                                unsigned short* __restrict__ dst, int n) {
  int i = blockIdx.x * blockDim.x + threadIdx.x;
  if (i < n) dst[i] = f2bf(src[i]);
}

__global__ void zero_us_kernel(unsigned short* __restrict__ dst, int n) {
  int i = blockIdx.x * blockDim.x + threadIdx.x;
  if (i < n) dst[i] = 0;
}

__global__ void lengths_kernel(const unsigned char* __restrict__ mask,
                               int* __restrict__ len) {
  __shared__ int red[256];
  int b = blockIdx.x, s = 0;
  for (int t = threadIdx.x; t < L_; t += 256) s += mask[b * L_ + t] ? 1 : 0;
  red[threadIdx.x] = s; __syncthreads();
  for (int off = 128; off > 0; off >>= 1) {
    if ((int)threadIdx.x < off) red[threadIdx.x] += red[threadIdx.x + off];
    __syncthreads();
  }
  if (threadIdx.x == 0) len[b] = red[0];
}

__global__ void rmsnorm_bf16_kernel(const float* __restrict__ x,
                                    const float* __restrict__ w,
                                    unsigned short* __restrict__ out) {
  __shared__ float red[256];
  int row = blockIdx.x;
  const float* xr = x + (size_t)row * DM;
  float s = 0.f;
  for (int i = threadIdx.x; i < DM; i += 256) { float v = xr[i]; s += v * v; }
  red[threadIdx.x] = s; __syncthreads();
  for (int off = 128; off > 0; off >>= 1) {
    if ((int)threadIdx.x < off) red[threadIdx.x] += red[threadIdx.x + off];
    __syncthreads();
  }
  float scale = rsqrtf(red[0] / (float)DM + 1e-6f);
  for (int i = threadIdx.x; i < DM; i += 256)
    out[(size_t)row * DM + i] = f2bf(xr[i] * scale * w[i]);
}

template <typename T>
__global__ void flip_kernel(const T* __restrict__ in,
                            const unsigned char* __restrict__ mask,
                            const int* __restrict__ len,
                            T* __restrict__ out, int Dd) {
  long long i = (long long)blockIdx.x * blockDim.x + threadIdx.x;
  long long total = (long long)B_ * L_ * Dd;
  if (i >= total) return;
  int d = (int)(i % Dd);
  int t = (int)((i / Dd) % L_);
  int b = (int)(i / ((long long)Dd * L_));
  int rv = len[b] - 1 - t;
  rv = rv < 0 ? 0 : (rv > L_ - 1 ? L_ - 1 : rv);
  T v = in[((size_t)b * L_ + rv) * Dd + d];
  out[i] = mask[b * L_ + t] ? v : (T)0;
}

// depthwise causal conv (k=4) + bias + silu + mask over xs slice of proj
__global__ void conv_kernel(const float* __restrict__ proj,
                            const float* __restrict__ cw,
                            const float* __restrict__ cb,
                            const unsigned char* __restrict__ mask,
                            float* __restrict__ xs_conv) {
  int i = blockIdx.x * blockDim.x + threadIdx.x;
  if (i >= MROWS * DI) return;
  int c = i % DI, row = i / DI, t = row % L_, b = row / L_;
  float acc = cb[c];
  #pragma unroll
  for (int j = 0; j < KC; j++) {
    int tt = t - (KC - 1) + j;
    if (tt >= 0) {
      float m = mask[b * L_ + tt] ? 1.f : 0.f;
      float xv = proj[((size_t)b * L_ + tt) * DPROJP + DI + c] * m;
      acc += cw[c * KC + j] * xv;
    }
  }
  float mt = mask[b * L_ + t] ? 1.f : 0.f;
  xs_conv[i] = siluf(acc) * mt;
}

__global__ void dta_kernel(const float* __restrict__ proj,
                           const float* __restrict__ dt_bias,
                           const float* __restrict__ A_log,
                           const unsigned char* __restrict__ mask,
                           float* __restrict__ dtb, float* __restrict__ ab) {
  int i = blockIdx.x * blockDim.x + threadIdx.x;
  if (i >= MROWS * NH) return;
  int h = i % NH, row = i / NH, t = row % L_, b = row / L_;
  float v = proj[(size_t)row * DPROJP + 2 * DI + 2 * RK * DS + h] + dt_bias[h];
  float sp = v > 20.f ? v : log1pf(expf(v));
  sp *= (mask[b * L_ + t] ? 1.f : 0.f);
  dtb[i] = sp;
  ab[i] = expf(sp * (-expf(A_log[h])));
}

// SSD scan: one 64-thread block per (batch, head); thread p owns S[:,p] (64 f32
// regs). B/C row vectors for step t+1 are prefetched into double-buffered LDS
// with CDNA5 async global->LDS loads while step t computes; s_wait_asynccnt 2
// retires only the older pair (async loads complete in order).
__global__ __launch_bounds__(64)
void scan_kernel(const float* __restrict__ proj,
                 const float* __restrict__ xs_conv,
                 const float* __restrict__ dtb,
                 const float* __restrict__ ab,
                 const float* __restrict__ Dv,
                 float* __restrict__ y) {
  __shared__ float Bs[2][DS], Cs[2][DS];
  int b = blockIdx.x / NH, h = blockIdx.x % NH, r = h / (NH / RK);
  int p = threadIdx.x;
  float S[DS];
  #pragma unroll
  for (int n = 0; n < DS; n++) S[n] = 0.f;
  float Dh = Dv[h];
  const size_t base = (size_t)b * L_;

  {  // prime step 0 into buffer 0
    const float* pr = proj + base * DPROJP;
    async_ld_b32((unsigned)(size_t)&Bs[0][p], pr + 2 * DI + r * DS + p);
    async_ld_b32((unsigned)(size_t)&Cs[0][p], pr + 2 * DI + RK * DS + r * DS + p);
  }
  for (int t = 0; t < L_; t++) {
    int cur = t & 1;
    __syncthreads();  // all waves done reading buf[cur^1] (step t-1)
    if (t + 1 < L_) {
      const float* pr = proj + (base + t + 1) * DPROJP;
      async_ld_b32((unsigned)(size_t)&Bs[cur ^ 1][p], pr + 2 * DI + r * DS + p);
      async_ld_b32((unsigned)(size_t)&Cs[cur ^ 1][p],
                   pr + 2 * DI + RK * DS + r * DS + p);
      asm volatile("s_wait_asynccnt 0x2" ::: "memory");
    } else {
      asm volatile("s_wait_asynccnt 0x0" ::: "memory");
    }
    __syncthreads();  // buf[cur] visible to both waves
    size_t row = base + t;
    float at  = ab[row * NH + h];
    float dtt = dtb[row * NH + h];
    float xv  = xs_conv[row * DI + h * HD + p];
    float dx = dtt * xv, acc = 0.f;
    #pragma unroll
    for (int n = 0; n < DS; n++) {
      S[n] = at * S[n] + dx * Bs[cur][n];
      acc += Cs[cur][n] * S[n];
    }
    y[row * DI + h * HD + p] = acc + Dh * xv;
  }
}

__global__ void ygate_kernel(const float* __restrict__ proj,
                             const float* __restrict__ y,
                             unsigned short* __restrict__ ybf) {
  int i = blockIdx.x * blockDim.x + threadIdx.x;
  if (i >= MROWS * DI) return;
  int c = i % DI; size_t row = i / DI;
  float z = proj[row * DPROJP + c];
  ybf[i] = f2bf(y[i] * siluf(z));
}

__global__ void cat_kernel(const float* __restrict__ yf,
                           const float* __restrict__ yb,
                           unsigned short* __restrict__ cat) {
  int i = blockIdx.x * blockDim.x + threadIdx.x;
  if (i >= MROWS * 2 * DM) return;
  int c = i % (2 * DM); size_t row = i / (2 * DM);
  float v = (c < DM) ? yf[row * DM + c] : yb[row * DM + (c - DM)];
  cat[i] = f2bf(v);
}

__global__ void mix_kernel(const float* __restrict__ gl,
                           const float* __restrict__ yf,
                           const float* __restrict__ yb,
                           unsigned short* __restrict__ mixbf) {
  int i = blockIdx.x * blockDim.x + threadIdx.x;
  if (i >= MROWS * DM) return;
  float g = sigf(gl[i]);
  mixbf[i] = f2bf(g * yf[i] + (1.f - g) * yb[i]);
}

__global__ void ffact_kernel(const float* __restrict__ u1,
                             const float* __restrict__ u3,
                             unsigned short* __restrict__ ffbf) {
  int i = blockIdx.x * blockDim.x + threadIdx.x;
  if (i >= MROWS * DFF) return;
  ffbf[i] = f2bf(siluf(u1[i]) * u3[i]);
}

__global__ void maskout_kernel(float* __restrict__ out,
                               const unsigned char* __restrict__ mask) {
  int i = blockIdx.x * blockDim.x + threadIdx.x;
  if (i >= MROWS * DM) return;
  int row = i / DM;
  if (!mask[row]) out[i] = 0.f;
}

// ---------------------------------------------------------------- host
static inline unsigned gblocks(long long n, int bs) {
  return (unsigned)((n + bs - 1) / bs);
}

extern "C" void kernel_launch(void* const* d_in, const int* in_sizes, int n_in,
                              void* d_out, int out_size, void* d_ws, size_t ws_size,
                              hipStream_t stream) {
  (void)in_sizes; (void)n_in; (void)out_size; (void)ws_size;
  const float*         x       = (const float*)d_in[0];
  const unsigned char* mask    = (const unsigned char*)d_in[1];
  const float*         norm1_w = (const float*)d_in[2];
  const float*         norm2_w = (const float*)d_in[3];
  const float*         gate_w  = (const float*)d_in[4];
  const float*         outp_w  = (const float*)d_in[5];
  const float*         ffn_w1  = (const float*)d_in[6];
  const float*         ffn_w3  = (const float*)d_in[7];
  const float*         ffn_w2  = (const float*)d_in[8];
  const float* in_w[2]    = { (const float*)d_in[9],  (const float*)d_in[16] };
  const float* conv_w[2]  = { (const float*)d_in[10], (const float*)d_in[17] };
  const float* conv_b[2]  = { (const float*)d_in[11], (const float*)d_in[18] };
  const float* dt_bias[2] = { (const float*)d_in[12], (const float*)d_in[19] };
  const float* A_log[2]   = { (const float*)d_in[13], (const float*)d_in[20] };
  const float* Dvec[2]    = { (const float*)d_in[14], (const float*)d_in[21] };
  const float* out_w[2]   = { (const float*)d_in[15], (const float*)d_in[22] };
  float* out = (float*)d_out;

  // workspace arena
  char* p = (char*)d_ws;
  auto alloc = [&](size_t bytes) -> void* {
    void* r = (void*)p;
    p += (bytes + 255) & ~(size_t)255;
    return r;
  };
  unsigned short* w_in_bf[2]  = { (unsigned short*)alloc((size_t)DPROJP * DM * 2),
                                  (unsigned short*)alloc((size_t)DPROJP * DM * 2) };
  unsigned short* w_out_bf[2] = { (unsigned short*)alloc((size_t)DM * DI * 2),
                                  (unsigned short*)alloc((size_t)DM * DI * 2) };
  unsigned short* w_gate_bf = (unsigned short*)alloc((size_t)DM * DI * 2);
  unsigned short* w_outp_bf = (unsigned short*)alloc((size_t)DM * DM * 2);
  unsigned short* w_ff1_bf  = (unsigned short*)alloc((size_t)DFF * DM * 2);
  unsigned short* w_ff3_bf  = (unsigned short*)alloc((size_t)DFF * DM * 2);
  unsigned short* w_ff2_bf  = (unsigned short*)alloc((size_t)DM * DFF * 2);
  unsigned short* h_bf    = (unsigned short*)alloc((size_t)MROWS * DM * 2);
  unsigned short* hrev_bf = (unsigned short*)alloc((size_t)MROWS * DM * 2);
  int*   lens    = (int*)alloc(B_ * sizeof(int));
  float* proj    = (float*)alloc((size_t)MROWS * DPROJP * 4);  // reused as u1 later
  float* xs_conv = (float*)alloc((size_t)MROWS * DI * 4);
  float* dtb     = (float*)alloc((size_t)MROWS * NH * 4);
  float* ab      = (float*)alloc((size_t)MROWS * NH * 4);
  float* ybuf    = (float*)alloc((size_t)MROWS * DI * 4);
  unsigned short* ybf = (unsigned short*)alloc((size_t)MROWS * DI * 2);
  float* yf    = (float*)alloc((size_t)MROWS * DM * 4);
  float* ybtmp = (float*)alloc((size_t)MROWS * DM * 4);
  float* yb    = (float*)alloc((size_t)MROWS * DM * 4);
  unsigned short* cat_bf = (unsigned short*)alloc((size_t)MROWS * DI * 2);
  float* gatel = (float*)alloc((size_t)MROWS * DM * 4);
  unsigned short* mix_bf = (unsigned short*)alloc((size_t)MROWS * DM * 2);
  float* x1    = (float*)alloc((size_t)MROWS * DM * 4);
  unsigned short* h2bf = (unsigned short*)alloc((size_t)MROWS * DM * 2);
  float* u1 = proj;  // reuse: proj no longer needed by the time FFN runs
  float* u3 = (float*)alloc((size_t)MROWS * DFF * 4);
  unsigned short* ffbf = (unsigned short*)alloc((size_t)MROWS * DFF * 2);

  const int BS = 256;
  auto cvt = [&](const float* s, unsigned short* d, long long n) {
    cvt_bf16_kernel<<<gblocks(n, BS), BS, 0, stream>>>(s, d, (int)n);
  };
  // weight conversions (f32 -> bf16); in-proj weight padded 4640 -> 4672 rows
  cvt(in_w[0],  w_in_bf[0],  (long long)DPROJ * DM);
  cvt(in_w[1],  w_in_bf[1],  (long long)DPROJ * DM);
  {
    long long padn = (long long)(DPROJP - DPROJ) * DM;   // 32*1024 zero rows
    zero_us_kernel<<<gblocks(padn, BS), BS, 0, stream>>>(
        w_in_bf[0] + (size_t)DPROJ * DM, (int)padn);
    zero_us_kernel<<<gblocks(padn, BS), BS, 0, stream>>>(
        w_in_bf[1] + (size_t)DPROJ * DM, (int)padn);
  }
  cvt(out_w[0], w_out_bf[0], (long long)DM * DI);
  cvt(out_w[1], w_out_bf[1], (long long)DM * DI);
  cvt(gate_w,   w_gate_bf,   (long long)DM * DI);
  cvt(outp_w,   w_outp_bf,   (long long)DM * DM);
  cvt(ffn_w1,   w_ff1_bf,    (long long)DFF * DM);
  cvt(ffn_w3,   w_ff3_bf,    (long long)DFF * DM);
  cvt(ffn_w2,   w_ff2_bf,    (long long)DM * DFF);

  lengths_kernel<<<B_, 256, 0, stream>>>(mask, lens);
  rmsnorm_bf16_kernel<<<MROWS, 256, 0, stream>>>(x, norm1_w, h_bf);
  flip_kernel<unsigned short><<<gblocks((long long)MROWS * DM, BS), BS, 0, stream>>>(
      h_bf, mask, lens, hrev_bf, DM);

  auto gemm = [&](const unsigned short* A, const unsigned short* W, float* Cc,
                  const float* resid, int N, int K) {
    dim3 g((unsigned)(N / 64), (unsigned)(MROWS / 256));
    if (resid)
      gemm_bf16_wmma<true ><<<g, 128, 0, stream>>>(A, W, Cc, resid, MROWS, N, K);
    else
      gemm_bf16_wmma<false><<<g, 128, 0, stream>>>(A, W, Cc, nullptr, MROWS, N, K);
  };

  for (int dir = 0; dir < 2; dir++) {
    gemm(dir ? hrev_bf : h_bf, w_in_bf[dir], proj, nullptr, DPROJP, DM);
    conv_kernel<<<gblocks((long long)MROWS * DI, BS), BS, 0, stream>>>(
        proj, conv_w[dir], conv_b[dir], mask, xs_conv);
    dta_kernel<<<gblocks((long long)MROWS * NH, BS), BS, 0, stream>>>(
        proj, dt_bias[dir], A_log[dir], mask, dtb, ab);
    scan_kernel<<<B_ * NH, 64, 0, stream>>>(proj, xs_conv, dtb, ab, Dvec[dir], ybuf);
    ygate_kernel<<<gblocks((long long)MROWS * DI, BS), BS, 0, stream>>>(proj, ybuf, ybf);
    gemm(ybf, w_out_bf[dir], dir ? ybtmp : yf, nullptr, DM, DI);
  }
  flip_kernel<float><<<gblocks((long long)MROWS * DM, BS), BS, 0, stream>>>(
      ybtmp, mask, lens, yb, DM);

  cat_kernel<<<gblocks((long long)MROWS * DI, BS), BS, 0, stream>>>(yf, yb, cat_bf);
  gemm(cat_bf, w_gate_bf, gatel, nullptr, DM, DI);
  mix_kernel<<<gblocks((long long)MROWS * DM, BS), BS, 0, stream>>>(gatel, yf, yb, mix_bf);
  gemm(mix_bf, w_outp_bf, x1, x, DM, DM);            // x1 = x + mix @ outp^T

  rmsnorm_bf16_kernel<<<MROWS, 256, 0, stream>>>(x1, norm2_w, h2bf);
  gemm(h2bf, w_ff1_bf, u1, nullptr, DFF, DM);
  gemm(h2bf, w_ff3_bf, u3, nullptr, DFF, DM);
  ffact_kernel<<<gblocks((long long)MROWS * DFF, BS), BS, 0, stream>>>(u1, u3, ffbf);
  gemm(ffbf, w_ff2_bf, out, x1, DM, DFF);            // out = x1 + ff @ w2^T
  maskout_kernel<<<gblocks((long long)MROWS * DM, BS), BS, 0, stream>>>(out, mask);
}
